// GumbelCodebook_62594853372333
// MI455X (gfx1250) — matrix operand order repaired and checked by reading, not compiled, not run on hardware
//
#include <hip/hip_runtime.h>
#include <hip/hip_bf16.h>

#define B_ 32
#define N_ 64
#define D_ 256
#define C_ 8192

typedef __attribute__((ext_vector_type(16))) __bf16 v16bf;
typedef __attribute__((ext_vector_type(8)))  float  v8f;

static __device__ __forceinline__ v8f wmma_bf16(v16bf a, v16bf b, v8f c) {
    return __builtin_amdgcn_wmma_f32_16x16x32_bf16(false, a, false, b, (short)0, c,
                                                   false, false);
}

// Load a 16x32 bf16 A-fragment whose K values are contiguous in memory.
// Layout: lane half hi=lane>>4; elems 0-15 <- K = hi*16+0..15
// (contiguous order).  Both runs are 16B-aligned float4 pairs.
static __device__ __forceinline__ v16bf load_a_contig(const float* __restrict__ base,
                                                      int hi) {
    const float4* p0 = (const float4*)(base + hi * 16);       // K = hi*16 ..
    const float4* p1 = (const float4*)(base + hi * 16 + 8);   // K = hi*16+8 ..
    float4 f0 = p0[0], f1 = p0[1], f2 = p1[0], f3 = p1[1];
    v16bf a;
    a[0]=(__bf16)f0.x; a[1]=(__bf16)f0.y; a[2]=(__bf16)f0.z; a[3]=(__bf16)f0.w;
    a[4]=(__bf16)f1.x; a[5]=(__bf16)f1.y; a[6]=(__bf16)f1.z; a[7]=(__bf16)f1.w;
    a[8]=(__bf16)f2.x; a[9]=(__bf16)f2.y; a[10]=(__bf16)f2.z; a[11]=(__bf16)f2.w;
    a[12]=(__bf16)f3.x; a[13]=(__bf16)f3.y; a[14]=(__bf16)f3.z; a[15]=(__bf16)f3.w;
    return a;
}

// Load a 32x16 bf16 B-fragment whose 16 K values are contiguous per lane
// (K-major source). Layout: lanes 0-15 hold K 0..15, lanes 16-31 hold 16..31;
// caller passes base already offset by hi*16.
static __device__ __forceinline__ v16bf load_b_contig(const float* __restrict__ base) {
    const float4* p = (const float4*)base;
    float4 f0 = p[0], f1 = p[1], f2 = p[2], f3 = p[3];
    v16bf b;
    b[0]=(__bf16)f0.x; b[1]=(__bf16)f0.y; b[2]=(__bf16)f0.z; b[3]=(__bf16)f0.w;
    b[4]=(__bf16)f1.x; b[5]=(__bf16)f1.y; b[6]=(__bf16)f1.z; b[7]=(__bf16)f1.w;
    b[8]=(__bf16)f2.x; b[9]=(__bf16)f2.y; b[10]=(__bf16)f2.z; b[11]=(__bf16)f2.w;
    b[12]=(__bf16)f3.x; b[13]=(__bf16)f3.y; b[14]=(__bf16)f3.z; b[15]=(__bf16)f3.w;
    return b;
}

// ---------------------------------------------------------------------------
// Kernel 1: log_alpha[b,n,c] = sum_d logits[b,n,d] * head[n,d,c]
// grid (C/256, N), 256 threads (8 waves). Block covers all 32 batch rows so
// the 536 MB `head` tensor is streamed from HBM exactly once.
// ---------------------------------------------------------------------------
__global__ void __launch_bounds__(256) k_head_gemm(
    const float* __restrict__ logits, const float* __restrict__ head,
    float* __restrict__ out_la)
{
    const int lane  = threadIdx.x & 31;
    const int wave  = threadIdx.x >> 5;
    const int n     = blockIdx.y;
    const int c_wav = blockIdx.x * 256 + wave * 32;   // 2 C-tiles per wave
    const int row16 = lane & 15;
    const int hi    = lane >> 4;

    v8f acc[2][2] = {};
    for (int k = 0; k < D_; k += 32) {
        v16bf afr[2];
#pragma unroll
        for (int mt = 0; mt < 2; ++mt) {
            const int b = mt * 16 + row16;
            afr[mt] = load_a_contig(logits + ((size_t)b * N_ + n) * D_ + k, hi);
        }
        v16bf bfr[2];
#pragma unroll
        for (int ct = 0; ct < 2; ++ct) {
            // head is C-major in K (stride C) -> dword loads, coalesced across lanes
            const int c = c_wav + ct * 16 + row16;
            const float* bp = head + ((size_t)n * D_ + (k + hi * 16)) * C_ + c;
            v16bf b;
#pragma unroll
            for (int j = 0; j < 16; ++j)
                b[j] = (__bf16)bp[(size_t)j * C_];
            bfr[ct] = b;
        }
#pragma unroll
        for (int mt = 0; mt < 2; ++mt)
#pragma unroll
            for (int ct = 0; ct < 2; ++ct)
                acc[mt][ct] = wmma_bf16(afr[mt], bfr[ct], acc[mt][ct]);
    }
#pragma unroll
    for (int mt = 0; mt < 2; ++mt)
#pragma unroll
        for (int ct = 0; ct < 2; ++ct) {
            const int c = c_wav + ct * 16 + row16;
#pragma unroll
            for (int r = 0; r < 8; ++r) {
                const int b = mt * 16 + hi * 8 + r;     // D-matrix row mapping
                out_la[((size_t)b * N_ + n) * C_ + c] = acc[mt][ct][r];
            }
        }
}

// ---------------------------------------------------------------------------
// Kernel 2: z = softmax((log_alpha + gumbel)/tau) over C per (b,n) row.
// One block per row; 8 float4 (32 elems) per thread held in VGPRs -> single
// HBM read, b128 VMEM ops throughout (each wave load touches a contiguous 512B).
// ---------------------------------------------------------------------------
__global__ void __launch_bounds__(256) k_softmax(
    const float* __restrict__ la, const float* __restrict__ gumbel,
    const float* __restrict__ tau, float* __restrict__ z)
{
    const int row = blockIdx.x;           // b*N + n
    const int tid = threadIdx.x;
    const float invtau = 1.0f / tau[0];
    const float4* lar = (const float4*)(la + (size_t)row * C_);
    const float4* gr  = (const float4*)(gumbel + (size_t)row * C_);

    float4 s[8];
    float m = -3.402823466e+38f;
#pragma unroll
    for (int t = 0; t < 8; ++t) {
        const int i = t * 256 + tid;      // float4 index, coalesced
        float4 a = lar[i], g = gr[i];
        s[t].x = (a.x + g.x) * invtau;
        s[t].y = (a.y + g.y) * invtau;
        s[t].z = (a.z + g.z) * invtau;
        s[t].w = (a.w + g.w) * invtau;
        m = fmaxf(m, fmaxf(fmaxf(s[t].x, s[t].y), fmaxf(s[t].z, s[t].w)));
    }
    __shared__ float red[256];
    red[tid] = m;
    __syncthreads();
    for (int off = 128; off > 0; off >>= 1) {
        if (tid < off) red[tid] = fmaxf(red[tid], red[tid + off]);
        __syncthreads();
    }
    const float rmax = red[0];
    __syncthreads();

    float sum = 0.f;
#pragma unroll
    for (int t = 0; t < 8; ++t) {
        s[t].x = __expf(s[t].x - rmax);
        s[t].y = __expf(s[t].y - rmax);
        s[t].z = __expf(s[t].z - rmax);
        s[t].w = __expf(s[t].w - rmax);
        sum += (s[t].x + s[t].y) + (s[t].z + s[t].w);
    }
    red[tid] = sum;
    __syncthreads();
    for (int off = 128; off > 0; off >>= 1) {
        if (tid < off) red[tid] += red[tid + off];
        __syncthreads();
    }
    const float rinv = 1.0f / red[0];
    float4* zr = (float4*)(z + (size_t)row * C_);
#pragma unroll
    for (int t = 0; t < 8; ++t) {
        float4 o;
        o.x = s[t].x * rinv; o.y = s[t].y * rinv;
        o.z = s[t].z * rinv; o.w = s[t].w * rinv;
        zr[t * 256 + tid] = o;
    }
}

// ---------------------------------------------------------------------------
// Kernel 3 (fused): q = z @ W^T  (K=8192) staged to LDS (bf16), then
// quantized = q @ pos_map[n] (K=256). grid (2, N), 256 threads.
// Both stage-A operands are K-contiguous -> b128 loads.
// ---------------------------------------------------------------------------
__global__ void __launch_bounds__(256) k_codebook_posmap(
    const float* __restrict__ z, const float* __restrict__ cw,
    const float* __restrict__ pm, float* __restrict__ outq)
{
    __shared__ __bf16 qlds[16 * 256];
    const int lane  = threadIdx.x & 31;
    const int wave  = threadIdx.x >> 5;
    const int btile = blockIdx.x;
    const int n     = blockIdx.y;
    const int row16 = lane & 15;
    const int hi    = lane >> 4;
    const int dw    = wave * 32;          // 2 column tiles per wave

    // -------- stage A: q[16 x 256] = z_tile[16 x 8192] * cw^T --------------
    v8f acc[2] = {};
    const int b0 = btile * 16 + row16;
    const float* zrow = z + ((size_t)b0 * N_ + n) * C_;
    for (int k = 0; k < C_; k += 32) {
        v16bf afr = load_a_contig(zrow + k, hi);
#pragma unroll
        for (int ct = 0; ct < 2; ++ct) {
            const int d = dw + ct * 16 + row16;
            v16bf bfr = load_b_contig(cw + (size_t)d * C_ + k + hi * 16);
            acc[ct] = wmma_bf16(afr, bfr, acc[ct]);
        }
    }
#pragma unroll
    for (int ct = 0; ct < 2; ++ct) {
        const int d = dw + ct * 16 + row16;
#pragma unroll
        for (int r = 0; r < 8; ++r)
            qlds[(hi * 8 + r) * 256 + d] = (__bf16)acc[ct][r];
    }
    __syncthreads();

    // -------- stage B: out[16 x 256] = q[16 x 256] * pm[n] -----------------
    v8f acc2[2] = {};
    for (int k = 0; k < D_; k += 32) {
        // A from LDS: 16 contiguous bf16 per lane (K = hi*16..)
        v16bf afr;
        const __bf16* qp0 = &qlds[row16 * 256 + k + hi * 16];
        const __bf16* qp1 = qp0 + 8;
#pragma unroll
        for (int j = 0; j < 8; ++j) { afr[j] = qp0[j]; afr[8 + j] = qp1[j]; }
#pragma unroll
        for (int ct = 0; ct < 2; ++ct) {
            const int mcol = dw + ct * 16 + row16;
            const float* bp = pm + (size_t)n * D_ * D_ + (size_t)(k + hi * 16) * D_ + mcol;
            v16bf bfr;
#pragma unroll
            for (int j = 0; j < 16; ++j)
                bfr[j] = (__bf16)bp[(size_t)j * D_];
            acc2[ct] = wmma_bf16(afr, bfr, acc2[ct]);
        }
    }
#pragma unroll
    for (int ct = 0; ct < 2; ++ct) {
        const int mcol = dw + ct * 16 + row16;
#pragma unroll
        for (int r = 0; r < 8; ++r) {
            const int b = btile * 16 + hi * 8 + r;
            outq[((size_t)b * N_ + n) * D_ + mcol] = acc2[ct][r];
        }
    }
}

extern "C" void kernel_launch(void* const* d_in, const int* in_sizes, int n_in,
                              void* d_out, int out_size, void* d_ws, size_t ws_size,
                              hipStream_t stream) {
    (void)in_sizes; (void)n_in; (void)out_size; (void)d_ws; (void)ws_size;
    const float* logits  = (const float*)d_in[0];   // [B,N,D]
    const float* head    = (const float*)d_in[1];   // [N,D,C]
    const float* pos_map = (const float*)d_in[2];   // [N,D,D]
    const float* cw      = (const float*)d_in[3];   // [D,C]
    const float* gumbel  = (const float*)d_in[4];   // [B,N,C]
    const float* tau     = (const float*)d_in[5];   // [1]

    float* out    = (float*)d_out;
    float* out_q  = out;                                  // B*N*D
    float* out_la = out + (size_t)B_ * N_ * D_;           // B*N*C
    float* out_z  = out_la + (size_t)B_ * N_ * C_;        // B*N*C

    k_head_gemm<<<dim3(C_ / 256, N_), 256, 0, stream>>>(logits, head, out_la);
    k_softmax<<<dim3(B_ * N_), 256, 0, stream>>>(out_la, gumbel, tau, out_z);
    k_codebook_posmap<<<dim3(2, N_), 256, 0, stream>>>(out_z, cw, pos_map, out_q);
}
